// UltraNet_MixQ_3238405341691
// MI455X (gfx1250) — matrix operand order repaired and checked by reading, not compiled, not run on hardware
//
#include <hip/hip_runtime.h>
#include <cstdint>
#include <cstddef>

// ---------------------------------------------------------------------------
// CDNA5 (gfx1250, wave32) mixed-precision QAT UltraNet + YOLO.
// Convs: implicit-GEMM over v_wmma_f32_16x16x32_f16, async LDS weight staging.
// ---------------------------------------------------------------------------

typedef __attribute__((ext_vector_type(16))) _Float16 v16h;
typedef __attribute__((ext_vector_type(4)))  _Float16 v4h;
typedef __attribute__((ext_vector_type(8)))  float    v8f;
typedef int i32x4 __attribute__((vector_size(16)));
typedef __attribute__((address_space(1))) i32x4 gl_i32x4;
typedef __attribute__((address_space(3))) i32x4 ld_i32x4;

#if defined(__has_builtin)
#  if __has_builtin(__builtin_amdgcn_global_load_async_to_lds_b128)
#    define HAVE_ASYNC_LDS 1
#  endif
#  if __has_builtin(__builtin_amdgcn_s_wait_asynccnt)
#    define HAVE_WAIT_ASYNC 1
#  endif
#endif
#ifndef HAVE_ASYNC_LDS
#  define HAVE_ASYNC_LDS 0
#endif
#ifndef HAVE_WAIT_ASYNC
#  define HAVE_WAIT_ASYNC 0
#endif

#define NLEV 7
__device__ __forceinline__ float lev(int j) {
    return (float)((1 << (j + 2)) - 1);   // 3,7,15,31,63,127,255
}

// ---------------------------------------------------------------------------
// 1) softmax over the 9x7 alpha rows
// ---------------------------------------------------------------------------
__global__ void softmax_rows_kernel(const float* __restrict__ aa,
                                    const float* __restrict__ aw,
                                    float* __restrict__ swa,
                                    float* __restrict__ sww) {
    int t = threadIdx.x;
    if (t >= 18) return;
    const float* src = (t < 9) ? (aa + t * NLEV) : (aw + (t - 9) * NLEV);
    float*       dst = (t < 9) ? (swa + t * NLEV) : (sww + (t - 9) * NLEV);
    float mx = -1e30f;
    for (int j = 0; j < NLEV; ++j) mx = fmaxf(mx, src[j]);
    float e[NLEV], s = 0.f;
    for (int j = 0; j < NLEV; ++j) { e[j] = expf(src[j] - mx); s += e[j]; }
    float inv = 1.f / s;
    for (int j = 0; j < NLEV; ++j) dst[j] = e[j] * inv;
}

// ---------------------------------------------------------------------------
// 2) per-layer max(|tanh(w)|), one block per layer
// ---------------------------------------------------------------------------
__global__ void wabsmax_kernel(const float* __restrict__ w0,
                               const float* __restrict__ w1,
                               const float* __restrict__ w2,
                               const float* __restrict__ w37,
                               const float* __restrict__ w8,
                               float* __restrict__ wmax) {
    __shared__ float red[256];
    int l = blockIdx.x;
    const float* w; int n;
    switch (l) {
        case 0: w = w0; n = 16 * 3 * 9;   break;
        case 1: w = w1; n = 32 * 16 * 9;  break;
        case 2: w = w2; n = 64 * 32 * 9;  break;
        case 8: w = w8; n = 36 * 64;      break;
        default: w = w37 + (size_t)(l - 3) * 64 * 64 * 9; n = 64 * 64 * 9; break;
    }
    float m = 0.f;
    for (int i = threadIdx.x; i < n; i += 256) m = fmaxf(m, fabsf(tanhf(w[i])));
    red[threadIdx.x] = m;
    __syncthreads();
    for (int s = 128; s > 0; s >>= 1) {
        if (threadIdx.x < s) red[threadIdx.x] = fmaxf(red[threadIdx.x], red[threadIdx.x + s]);
        __syncthreads();
    }
    if (threadIdx.x == 0) wmax[l] = red[0];
}

// ---------------------------------------------------------------------------
// 3) quantize + pack weights into WMMA A-fragment layout (f16)
//    dst linear index = ((mt*ktiles + kt)*32 + lane)*16 + half
// ---------------------------------------------------------------------------
__global__ void pack_weights_kernel(const float* __restrict__ w,
                                    const float* __restrict__ wmax,
                                    const float* __restrict__ sw,
                                    _Float16* __restrict__ dst,
                                    int OC, int IC, int ks, int mtiles, int ktiles) {
    int idx = blockIdx.x * blockDim.x + threadIdx.x;
    int total = mtiles * ktiles * 512;
    if (idx >= total) return;
    int half = idx & 15;
    int lane = (idx >> 4) & 31;
    int tile = idx >> 9;
    int kt = tile % ktiles;
    int mt = tile / ktiles;

    int m_local, k_in;
    if (lane < 16) { m_local = lane;      k_in = half;  }
    else           { m_local = lane - 16; k_in = half + 16; }
    int oc = mt * 16 + m_local;
    int k  = kt * 32 + k_in;
    int ks2 = ks * ks;
    int K = IC * ks2;

    float val = 0.f;
    if (oc < OC && k < K) {
        int ic = k / ks2;
        int r2 = k - ic * ks2;
        int kh = r2 / ks, kw = r2 - kh * ks;
        float wv = w[(((size_t)oc * IC + ic) * ks + kh) * ks + kw];
        float wt = tanhf(wv) / wmax[0];
        float wn = (wt + 1.f) * 0.5f;
        float q = 0.f;
        #pragma unroll
        for (int j = 0; j < NLEV; ++j) {
            float nj = lev(j);
            q += sw[j] * (rintf(wn * nj) / nj);
        }
        val = 2.f * q - 1.f;
    }
    dst[idx] = (_Float16)val;
}

// ---------------------------------------------------------------------------
// 4) activation mixed quantization, f32 -> f16, 4-wide
// ---------------------------------------------------------------------------
__global__ void quant_act_kernel(const float4* __restrict__ src,
                                 v4h* __restrict__ dst,
                                 long n4, const float* __restrict__ sw) {
    float s[NLEV];
    #pragma unroll
    for (int j = 0; j < NLEV; ++j) s[j] = sw[j];
    long stride = (long)gridDim.x * blockDim.x;
    for (long i = (long)blockIdx.x * blockDim.x + threadIdx.x; i < n4; i += stride) {
        float4 v = src[i];
        float in[4] = {v.x, v.y, v.z, v.w};
        v4h o;
        #pragma unroll
        for (int c = 0; c < 4; ++c) {
            float xc = fminf(fmaxf(in[c], 0.f), 1.f);
            float q = 0.f;
            #pragma unroll
            for (int j = 0; j < NLEV; ++j) {
                float nj = lev(j);
                q += s[j] * (rintf(xc * nj) / nj);
            }
            o[c] = (_Float16)q;
        }
        dst[i] = o;
    }
}

// ---------------------------------------------------------------------------
// 5) implicit-GEMM conv with WMMA, fused BN epilogue.
//    Block = 256 threads = 8 waves. Block tile: 16 OC x 256 N.
//    Each wave: 16 OC x 32 N -> 2 accumulators, 2 wmma per K-step.
//    Weight K-tile staged in LDS via GLOBAL_LOAD_ASYNC_TO_LDS_B128.
// ---------------------------------------------------------------------------
#define BROW 48   // padded B-tile row stride in halfs (96B, keeps 32B alignment)

template <int KS_>
__global__ __launch_bounds__(256)
void conv_wmma_kernel(const _Float16* __restrict__ act,
                      const _Float16* __restrict__ wpk,
                      float* __restrict__ out,
                      int IC, int H, int W, int OC,
                      int pad, int ktiles,
                      const float* __restrict__ bng, const float* __restrict__ bnb,
                      const float* __restrict__ bnm, const float* __restrict__ bnv) {
    __shared__ alignas(64) _Float16 btile[256 * BROW];   // 24 KB
    __shared__ alignas(64) _Float16 abuf[512];           // 1 KB weight K-tile

    const int tid  = threadIdx.x;
    const int wv   = tid >> 5;
    const int lane = tid & 31;
    const int HW   = H * W;
    const int N    = 16 * HW;                 // batch = 16
    const int n_base = blockIdx.x * 256;
    const int mtile  = blockIdx.y;
    const int ks2 = KS_ * KS_;
    const int K   = IC * ks2;

    // im2col fill: one row (32 k-values) per thread
    const int fn = tid;
    int fb = 0, foh = 0, fow = 0;
    bool fvalid = false;
    {
        int n = n_base + fn;
        if (n < N) {
            fvalid = true;
            fb = n / HW;
            int r = n - fb * HW;
            foh = r / W;
            fow = r - foh * W;
        }
    }

    v8f acc0 = {};
    v8f acc1 = {};
    for (int kt = 0; kt < ktiles; ++kt) {
        __syncthreads();

        // ---- stage weight A-tile (1 KB) into LDS, wave 0 only ----
        if (tid < 32) {
            const _Float16* ap = wpk + (((size_t)mtile * ktiles + kt) << 9) + lane * 16;
#if HAVE_ASYNC_LDS
            gl_i32x4* gsrc = (gl_i32x4*)(const_cast<_Float16*>(ap));
            ld_i32x4* ldst = (ld_i32x4*)(&abuf[lane * 16]);
            __builtin_amdgcn_global_load_async_to_lds_b128(gsrc, ldst, 0, 0);
            __builtin_amdgcn_global_load_async_to_lds_b128(gsrc, ldst, 16, 0);
#else
            *(v16h*)&abuf[lane * 16] = *(const v16h*)ap;
#endif
        }

        // ---- im2col fill of B tile (overlaps with async A copy) ----
        _Float16 tmp[32];
        #pragma unroll
        for (int i = 0; i < 32; ++i) {
            int kk = kt * 32 + i;
            _Float16 hv = (_Float16)0.f;
            if (fvalid && kk < K) {
                int ic = kk / ks2;
                int r2 = kk - ic * ks2;
                int kh = r2 / KS_, kw = r2 - kh * KS_;
                int ih = foh + kh - pad;
                int iw = fow + kw - pad;
                if (ih >= 0 && ih < H && iw >= 0 && iw < W)
                    hv = act[(((size_t)fb * IC + ic) * H + ih) * W + iw];
            }
            tmp[i] = hv;
        }
        *(v16h*)&btile[fn * BROW]      = *(v16h*)&tmp[0];
        *(v16h*)&btile[fn * BROW + 16] = *(v16h*)&tmp[16];

#if HAVE_ASYNC_LDS
        if (tid < 32) {
#if HAVE_WAIT_ASYNC
            __builtin_amdgcn_s_wait_asynccnt(0);
#else
            asm volatile("s_wait_asynccnt 0" ::: "memory");
#endif
        }
#endif
        __syncthreads();

        // ---- fragments + 2x WMMA ----
        v16h afrag = *(const v16h*)&abuf[lane * 16];
        const int nloc = wv * 32 + (lane & 15);
        const int koff = (lane < 16) ? 0 : 16;
        v16h bfrag0 = *(const v16h*)&btile[nloc * BROW + koff];
        v16h bfrag1 = *(const v16h*)&btile[(nloc + 16) * BROW + koff];

        acc0 = __builtin_amdgcn_wmma_f32_16x16x32_f16(
            false, afrag, false, bfrag0, (short)0, acc0, false, false);
        acc1 = __builtin_amdgcn_wmma_f32_16x16x32_f16(
            false, afrag, false, bfrag1, (short)0, acc1, false, false);
    }

    // ---- epilogue: C/D map: reg r -> M = r (+8 for lanes>=16), N = lane&15 ----
    const int ocBase = mtile * 16 + ((lane < 16) ? 0 : 8);
    #pragma unroll
    for (int half = 0; half < 2; ++half) {
        int n = n_base + wv * 32 + (lane & 15) + half * 16;
        if (n < N) {
            int b = n / HW;
            int r = n - b * HW;
            int oh = r / W;
            int ow = r - oh * W;
            const v8f& acc = half ? acc1 : acc0;
            #pragma unroll
            for (int rI = 0; rI < 8; ++rI) {
                int oc = ocBase + rI;
                if (oc < OC) {
                    float v = acc[rI];
                    if (bng != nullptr) {
                        float s = bng[oc] * rsqrtf(bnv[oc] + 1e-5f);
                        v = v * s + (bnb[oc] - bnm[oc] * s);
                    }
                    out[(((size_t)b * OC + oc) * H + oh) * W + ow] = v;
                }
            }
        }
    }
}

// ---------------------------------------------------------------------------
// 6) 2x2 max-pool
// ---------------------------------------------------------------------------
__global__ void pool2x2_kernel(const float* __restrict__ in, float* __restrict__ out,
                               long BC, int H, int W) {
    int oH = H >> 1, oW = W >> 1;
    long total = BC * oH * oW;
    long stride = (long)gridDim.x * blockDim.x;
    for (long idx = (long)blockIdx.x * blockDim.x + threadIdx.x; idx < total; idx += stride) {
        int ow = (int)(idx % oW);
        long t = idx / oW;
        int oh = (int)(t % oH);
        long c = t / oH;
        size_t base = ((size_t)c * H + (size_t)oh * 2) * W + (size_t)ow * 2;
        float m = fmaxf(fmaxf(in[base], in[base + 1]),
                        fmaxf(in[base + W], in[base + W + 1]));
        out[idx] = m;
    }
}

// ---------------------------------------------------------------------------
// 7) YOLO decode: raw (16,36,16,32) -> io (16,3072,6) ++ p (16,6,16,32,6)
// ---------------------------------------------------------------------------
__global__ void yolo_kernel(const float* __restrict__ raw, float* __restrict__ out) {
    const int total = 16 * 6 * 16 * 32;
    int idx = blockIdx.x * blockDim.x + threadIdx.x;
    if (idx >= total) return;
    int x = idx & 31;
    int t = idx >> 5;
    int y = t & 15;
    t >>= 4;
    int a = t % 6;
    int bs = t / 6;

    float p[6];
    #pragma unroll
    for (int o = 0; o < 6; ++o)
        p[o] = raw[(((size_t)bs * 36 + a * 6 + o) * 16 + y) * 32 + x];

    size_t base = (size_t)idx * 6;
    const size_t P_OFF = (size_t)16 * 3072 * 6;
    #pragma unroll
    for (int o = 0; o < 6; ++o) out[P_OFF + base + o] = p[o];

    const float stride = 16.f;       // 512 / 32
    const float anchor_s = 20.f;     // 1.25 * 16
    float s0 = 1.f / (1.f + expf(-p[0]));
    float s1 = 1.f / (1.f + expf(-p[1]));
    out[base + 0] = (s0 + (float)x) * stride;
    out[base + 1] = (s1 + (float)y) * stride;
    out[base + 2] = expf(p[2]) * anchor_s;
    out[base + 3] = expf(p[3]) * anchor_s;
    out[base + 4] = 1.f / (1.f + expf(-p[4]));
    out[base + 5] = 1.f / (1.f + expf(-p[5]));
}

// ---------------------------------------------------------------------------
// host side
// ---------------------------------------------------------------------------
extern "C" void kernel_launch(void* const* d_in, const int* in_sizes, int n_in,
                              void* d_out, int out_size, void* d_ws, size_t ws_size,
                              hipStream_t stream) {
    const float* x   = (const float*)d_in[0];
    const float* w0  = (const float*)d_in[1];
    const float* w1  = (const float*)d_in[2];
    const float* w2  = (const float*)d_in[3];
    const float* w37 = (const float*)d_in[4];
    const float* w8  = (const float*)d_in[5];
    const float* aa  = (const float*)d_in[6];
    const float* aw  = (const float*)d_in[7];
    const float* g0  = (const float*)d_in[8];
    const float* b0  = (const float*)d_in[9];
    const float* m0  = (const float*)d_in[10];
    const float* v0  = (const float*)d_in[11];
    const float* g1  = (const float*)d_in[12];
    const float* b1  = (const float*)d_in[13];
    const float* m1  = (const float*)d_in[14];
    const float* v1  = (const float*)d_in[15];
    const float* g27 = (const float*)d_in[16];
    const float* b27 = (const float*)d_in[17];
    const float* m27 = (const float*)d_in[18];
    const float* v27 = (const float*)d_in[19];

    uint8_t* ws = (uint8_t*)d_ws;
    float*     swa  = (float*)(ws + 0);
    float*     sww  = (float*)(ws + 512);
    float*     wmax = (float*)(ws + 1024);
    _Float16*  wpk  = (_Float16*)(ws + 4096);
    _Float16*  actq = (_Float16*)(ws + (size_t)1  * 1048576);   // <= 34 MB
    float*     convb= (float*)(ws + (size_t)36 * 1048576);      // <= 135 MB
    float*     poolb= (float*)(ws + (size_t)176 * 1048576);     // <= 68 MB

    static const int IC[9] = {3, 16, 32, 64, 64, 64, 64, 64, 64};
    static const int OC[9] = {16, 32, 64, 64, 64, 64, 64, 64, 36};
    static const int HH[9] = {256, 128, 64, 32, 16, 16, 16, 16, 16};
    static const int WW[9] = {512, 256, 128, 64, 32, 32, 32, 32, 32};
    static const int KS[9] = {3, 3, 3, 3, 3, 3, 3, 3, 1};
    static const int PD[9] = {1, 1, 1, 1, 1, 1, 1, 1, 0};
    static const int MT[9] = {1, 2, 4, 4, 4, 4, 4, 4, 3};
    static const int KT[9] = {1, 5, 9, 18, 18, 18, 18, 18, 2};
    static const size_t WOFF[9] = {0, 512, 5632, 24064, 60928, 97792, 134656, 171520, 208384};

    const float* wptr[9] = {
        w0, w1, w2,
        w37 + (size_t)0 * 36864, w37 + (size_t)1 * 36864, w37 + (size_t)2 * 36864,
        w37 + (size_t)3 * 36864, w37 + (size_t)4 * 36864, w8
    };
    const float* bng[9] = {g0, g1, g27 + 0, g27 + 64, g27 + 128, g27 + 192, g27 + 256, g27 + 320, nullptr};
    const float* bnb[9] = {b0, b1, b27 + 0, b27 + 64, b27 + 128, b27 + 192, b27 + 256, b27 + 320, nullptr};
    const float* bnm[9] = {m0, m1, m27 + 0, m27 + 64, m27 + 128, m27 + 192, m27 + 256, m27 + 320, nullptr};
    const float* bnv[9] = {v0, v1, v27 + 0, v27 + 64, v27 + 128, v27 + 192, v27 + 256, v27 + 320, nullptr};

    // --- setup ---
    softmax_rows_kernel<<<1, 64, 0, stream>>>(aa, aw, swa, sww);
    wabsmax_kernel<<<9, 256, 0, stream>>>(w0, w1, w2, w37, w8, wmax);
    for (int l = 0; l < 9; ++l) {
        int total = MT[l] * KT[l] * 512;
        pack_weights_kernel<<<(total + 255) / 256, 256, 0, stream>>>(
            wptr[l], wmax + l, sww + l * NLEV, wpk + WOFF[l],
            OC[l], IC[l], KS[l], MT[l], KT[l]);
    }

    // --- layer pipeline ---
    const float* cur = x;
    for (int l = 0; l < 9; ++l) {
        long cnt = (long)16 * IC[l] * HH[l] * WW[l];
        long n4 = cnt >> 2;
        int qb = (int)((n4 + 255) / 256);
        if (qb > 4096) qb = 4096;
        quant_act_kernel<<<qb, 256, 0, stream>>>(
            (const float4*)cur, (v4h*)actq, n4, swa + l * NLEV);

        int N = 16 * HH[l] * WW[l];
        dim3 grid((N + 255) / 256, MT[l]);
        if (KS[l] == 3) {
            conv_wmma_kernel<3><<<grid, 256, 0, stream>>>(
                actq, wpk + WOFF[l], convb,
                IC[l], HH[l], WW[l], OC[l], PD[l], KT[l],
                bng[l], bnb[l], bnm[l], bnv[l]);
        } else {
            conv_wmma_kernel<1><<<grid, 256, 0, stream>>>(
                actq, wpk + WOFF[l], convb,
                IC[l], HH[l], WW[l], OC[l], PD[l], KT[l],
                bng[l], bnb[l], bnm[l], bnv[l]);
        }

        if (l < 4) {
            long BC = (long)16 * OC[l];
            long total = BC * (HH[l] / 2) * (WW[l] / 2);
            int pb = (int)((total + 1023) / 1024);
            if (pb > 8192) pb = 8192;
            pool2x2_kernel<<<pb, 256, 0, stream>>>(convb, poolb, BC, HH[l], WW[l]);
            cur = poolb;
        } else {
            cur = convb;
        }
    }

    // --- YOLO decode ---
    const int ytotal = 16 * 6 * 16 * 32;
    yolo_kernel<<<(ytotal + 255) / 256, 256, 0, stream>>>(convb, (float*)d_out);
}
